// MulticlassHausdorff_86852828660213
// MI455X (gfx1250) — compile-verified, matrix operand order checked by reading, not run on hardware
//
#include <hip/hip_runtime.h>
#include <hip/hip_bf16.h>
#include <cstdint>
#include <math.h>

typedef __attribute__((ext_vector_type(16))) _Float16 v16h;
typedef __attribute__((ext_vector_type(8)))  float    v8f;

#define HH 256
#define WW 256
#define NPIX 65536
#define NCLS 26
#define NB 2
#define NWORDS 2048              // 256x256 bits / 32
#define GINF 0x10000000u         // "no edge in this column" sentinel (adds stay < 2^31)

struct Smem {
  uint32_t A[NPIX];              // 256 KB: g (squared col-dist) then dt^2, in place
  uint32_t pe[NWORDS];           // 8 KB: pred boundary bitmap
  uint32_t te[NWORDS];           // 8 KB: target boundary bitmap
  uint32_t hist[512];            // 2 KB: coarse / fine histograms
  uint32_t rowbuf[WW];           // 1 KB: row staging for in-place pass 2
  float    cnt[256];             // per-thread popcount partials (WMMA input)
  float    fbuf[40];             // WMMA lane sums + broadcast slot
  int      ibuf[8];              // integer broadcast (bins / ranks / selected dt^2)
};

static __device__ __forceinline__ uint32_t umin32(uint32_t a, uint32_t b) { return a < b ? a : b; }

// ---------------------------------------------------------------------------
// Kernel 1: argmax labels for predicted and target (softmax is monotone).
// ---------------------------------------------------------------------------
__global__ void labels_kernel(const float* __restrict__ pred,
                              const float* __restrict__ targ,
                              uint8_t* __restrict__ pl, uint8_t* __restrict__ tl) {
  int g = blockIdx.x * blockDim.x + threadIdx.x;    // 0 .. NB*NPIX-1
  if (g >= NB * NPIX) return;
  int b = g >> 16;
  int p = g & (NPIX - 1);
  const float* pp = pred + (size_t)b * NCLS * NPIX + p;
  const float* tp = targ + (size_t)b * NCLS * NPIX + p;
  float bvp = pp[0]; int bip = 0;
  float bvt = tp[0]; int bit_ = 0;
  for (int c = 1; c < NCLS; ++c) {
    if (c + 1 < NCLS) {
      __builtin_prefetch(pp + (size_t)(c + 1) * NPIX, 0, 3);   // global_prefetch_b8
    }
    float v = pp[(size_t)c * NPIX];
    if (v > bvp) { bvp = v; bip = c; }                          // first-max, like jnp.argmax
    float u = tp[(size_t)c * NPIX];
    if (u > bvt) { bvt = u; bit_ = c; }
  }
  pl[g] = (uint8_t)bip;
  tl[g] = (uint8_t)bit_;
}

// ---------------------------------------------------------------------------
// Exact popcount of a 2048-word bitmap via v_wmma_f32_16x16x32_f16.
// 256 per-thread partials (each <= 256, exact in f16) fill A of wave 0,
// B = ones => D rows replicate row-sums; sum(D) = 16 * total (exact in f32).
// ---------------------------------------------------------------------------
static __device__ int count_bits_wmma(Smem& sm, const uint32_t* bm, int tid) {
  int partial = 0;
  for (int w = tid; w < NWORDS; w += 256) partial += __popc(bm[w]);
  sm.cnt[tid] = (float)partial;
  __syncthreads();
  if (tid < 32) {                      // wave 0 only; all 32 lanes active -> EXEC all ones
    v16h a, bb;
    for (int i = 0; i < 16; ++i) { a[i] = (_Float16)0.0f; bb[i] = (_Float16)1.0f; }
    for (int i = 0; i < 8; ++i)  a[i] = (_Float16)sm.cnt[tid * 8 + i];
    v8f cz;
    for (int i = 0; i < 8; ++i) cz[i] = 0.0f;
    v8f dres = __builtin_amdgcn_wmma_f32_16x16x32_f16(
        /*neg_a=*/false, a, /*neg_b=*/false, bb,
        /*c_mod=*/(short)0, cz, /*reuse_a=*/false, /*reuse_b=*/false);
    float s = 0.0f;
    for (int i = 0; i < 8; ++i) s += dres[i];
    sm.fbuf[tid] = s;
  }
  __syncthreads();
  if (tid == 0) {
    float t = 0.0f;
    for (int i = 0; i < 32; ++i) t += sm.fbuf[i];
    sm.fbuf[32] = t * (1.0f / 16.0f);          // undo x16 replication; exact
  }
  __syncthreads();
  int n = (int)(sm.fbuf[32] + 0.5f);
  __syncthreads();
  return n;
}

// ---------------------------------------------------------------------------
// Exact EDT^2 to edge bitmap E, result (integer dt^2) into sm.A.
// Pass 1: per-column exact 1D nearest-edge distance, squared.
// Pass 2: per-row min-plus with valid pruning (d^2 >= cur => stop), row-staged.
// ---------------------------------------------------------------------------
static __device__ void edt_to(Smem& sm, const uint32_t* E, int tid) {
  __syncthreads();
  int j = tid;                                   // one thread per column
  uint32_t d = 1023u;
  for (int i = 0; i < HH; ++i) {                 // downward sweep: dist to edge above
    uint32_t bit = (E[i * 8 + (j >> 5)] >> (j & 31)) & 1u;
    d = bit ? 0u : umin32(d + 1u, 1023u);
    sm.A[i * WW + j] = d;
  }
  d = 1023u;
  for (int i = HH - 1; i >= 0; --i) {            // upward sweep + square
    uint32_t bit = (E[i * 8 + (j >> 5)] >> (j & 31)) & 1u;
    d = bit ? 0u : umin32(d + 1u, 1023u);
    uint32_t g = umin32(sm.A[i * WW + j], d);
    sm.A[i * WW + j] = (g > 255u) ? GINF : g * g;
  }
  __syncthreads();
  for (int i = 0; i < HH; ++i) {                 // row pass, all threads on row i
    sm.rowbuf[j] = sm.A[i * WW + j];
    __syncthreads();
    uint32_t cur = sm.rowbuf[j];
    for (int dd = 1; dd < WW; ++dd) {
      uint32_t d2 = (uint32_t)(dd * dd);
      if (d2 >= cur) break;                      // valid bound: further terms >= d2
      int l = j - dd;
      if (l >= 0) { uint32_t v = sm.rowbuf[l] + d2; if (v < cur) cur = v; }
      int r = j + dd;
      if (r < WW) { uint32_t v = sm.rowbuf[r] + d2; if (v < cur) cur = v; }
    }
    sm.A[i * WW + j] = cur;                      // exact integer dt^2 (<= 130050)
    __syncthreads();
  }
}

// ---------------------------------------------------------------------------
// Masked 95th percentile (linear interpolation) of sqrt(A) over bitmap M.
// dt^2 is an exact integer -> exact rank selection via 2-level LDS histogram.
// ---------------------------------------------------------------------------
static __device__ float pct95(Smem& sm, const uint32_t* M, int n, int tid) {
  float idxf = 0.95f * (float)(n - 1);           // replicate f32 arithmetic of reference
  int lo = (int)floorf(idxf);
  int hi = (int)ceilf(idxf);
  for (int w = tid; w < 512; w += 256) sm.hist[w] = 0u;
  __syncthreads();
  for (int p = tid; p < NPIX; p += 256)
    if ((M[p >> 5] >> (p & 31)) & 1u)
      atomicAdd(&sm.hist[sm.A[p] >> 8], 1u);     // coarse: 512 bins of width 256
  __syncthreads();
  if (tid == 0) {
    uint32_t cum = 0; int bl = -1, rl = 0, bh = -1, rh = 0;
    for (int i = 0; i < 512; ++i) {
      uint32_t nc = cum + sm.hist[i];
      if (bl < 0 && (uint32_t)lo < nc) { bl = i; rl = lo - (int)cum; }
      if (bh < 0 && (uint32_t)hi < nc) { bh = i; rh = hi - (int)cum; }
      cum = nc;
    }
    sm.ibuf[0] = bl; sm.ibuf[1] = rl; sm.ibuf[2] = bh; sm.ibuf[3] = rh;
  }
  __syncthreads();
  int bl = sm.ibuf[0], rl = sm.ibuf[1], bh = sm.ibuf[2], rh = sm.ibuf[3];
  for (int w = tid; w < 512; w += 256) sm.hist[w] = 0u;
  __syncthreads();
  for (int p = tid; p < NPIX; p += 256)
    if ((M[p >> 5] >> (p & 31)) & 1u) {
      uint32_t v = sm.A[p];
      int bin = (int)(v >> 8);
      if (bin == bl) atomicAdd(&sm.hist[v & 255u], 1u);
      if (bin == bh) atomicAdd(&sm.hist[256u + (v & 255u)], 1u);
    }
  __syncthreads();
  if (tid == 0) {
    uint32_t cum = 0; int ol = 0;
    for (int i = 0; i < 256; ++i) { cum += sm.hist[i]; if ((uint32_t)rl < cum) { ol = i; break; } }
    cum = 0; int oh = 0;
    for (int i = 0; i < 256; ++i) { cum += sm.hist[256 + i]; if ((uint32_t)rh < cum) { oh = i; break; } }
    sm.ibuf[4] = bl * 256 + ol;
    sm.ibuf[5] = bh * 256 + oh;
  }
  __syncthreads();
  float vlo = sqrtf((float)sm.ibuf[4]);
  float vhi = sqrtf((float)sm.ibuf[5]);
  float r = vlo + (vhi - vlo) * (idxf - floorf(idxf));
  __syncthreads();                               // protect hist/ibuf reuse
  return r;
}

// ---------------------------------------------------------------------------
// Kernel 2: one workgroup per (class, batch) pair; everything in LDS.
// ---------------------------------------------------------------------------
__global__ void __launch_bounds__(256) hd_pair_kernel(const uint8_t* __restrict__ pl,
                                                      const uint8_t* __restrict__ tl,
                                                      float* __restrict__ hd_out,
                                                      float* __restrict__ pres_out) {
  extern __shared__ unsigned char smraw[];
  Smem& sm = *reinterpret_cast<Smem*>(smraw);
  const int tid = threadIdx.x;
  const int c = blockIdx.x;
  const int b = blockIdx.y;
  const uint32_t* pl32 = (const uint32_t*)(pl + (size_t)b * NPIX);
  const uint32_t* tl32 = (const uint32_t*)(tl + (size_t)b * NPIX);

  // Build class masks as bitmaps (pm in sm.pe, tm in sm.te).
  for (int s = 0; s < 8; ++s) {
    int w = tid + 256 * s;                        // bitmap word: 32 pixels
    uint32_t pmb = 0u, tmb = 0u;
    for (int m = 0; m < 8; ++m) {
      uint32_t q = pl32[w * 8 + m];
      uint32_t u = tl32[w * 8 + m];
      for (int e = 0; e < 4; ++e) {
        if (((q >> (8 * e)) & 255u) == (uint32_t)c) pmb |= 1u << (m * 4 + e);
        if (((u >> (8 * e)) & 255u) == (uint32_t)c) tmb |= 1u << (m * 4 + e);
      }
    }
    sm.pe[w] = pmb; sm.te[w] = tmb;
  }
  __syncthreads();

  // Boundary = m & ~erode(m), cross element, border_value = 0 (register-staged, in place).
  uint32_t bp[8], bt[8];
  for (int s = 0; s < 8; ++s) {
    int w = tid + 256 * s;
    int cw = w & 7;                               // word index within row
    for (int which = 0; which < 2; ++which) {
      const uint32_t* bm = which ? sm.te : sm.pe;
      uint32_t m   = bm[w];
      uint32_t no  = (w >= 8) ? bm[w - 8] : 0u;             // north row
      uint32_t so  = (w < NWORDS - 8) ? bm[w + 8] : 0u;     // south row
      uint32_t lft = cw ? bm[w - 1] : 0u;
      uint32_t wv  = (m << 1) | (lft >> 31);                // west-neighbor values
      uint32_t ev  = (m >> 1) | ((cw < 7) ? (bm[w + 1] << 31) : 0u);  // east
      uint32_t er  = m & no & so & wv & ev;
      if (which) bt[s] = m & ~er; else bp[s] = m & ~er;
    }
  }
  __syncthreads();
  for (int s = 0; s < 8; ++s) { int w = tid + 256 * s; sm.pe[w] = bp[s]; sm.te[w] = bt[s]; }
  __syncthreads();

  int n_pe = count_bits_wmma(sm, sm.pe, tid);     // WMMA-based exact popcount
  int n_te = count_bits_wmma(sm, sm.te, tid);
  int oidx = c * NB + b;

  if (n_te == 0) {                                // class absent in target (pres=false)
    if (tid == 0) { hd_out[oidx] = 0.0f; pres_out[oidx] = 0.0f; }
    return;
  }
  if (n_pe == 0) {                                // present in target, pred misses -> NaN
    if (tid == 0) { hd_out[oidx] = __builtin_nanf(""); pres_out[oidx] = 1.0f; }
    return;
  }

  edt_to(sm, sm.te, tid);                         // dist^2 to target boundary
  float d_pt = pct95(sm, sm.pe, n_pe, tid);       // pred-edge -> target-edge
  edt_to(sm, sm.pe, tid);                         // dist^2 to pred boundary
  float d_tp = pct95(sm, sm.te, n_te, tid);       // target-edge -> pred-edge

  if (tid == 0) { hd_out[oidx] = fmaxf(d_pt, d_tp); pres_out[oidx] = 1.0f; }
}

// ---------------------------------------------------------------------------
// Kernel 3: presence-weighted means and final scalar.
// ---------------------------------------------------------------------------
__global__ void finalize_kernel(const float* __restrict__ hd,
                                const float* __restrict__ pres,
                                float* __restrict__ out) {
  if (threadIdx.x == 0 && blockIdx.x == 0) {
    float tot = 0.0f; int nv = 0;
    for (int b = 0; b < NB; ++b) {
      float num = 0.0f, den = 0.0f;
      for (int c = 0; c < NCLS; ++c) {
        float pr = pres[c * NB + b];
        num += (pr > 0.5f) ? hd[c * NB + b] : 0.0f;   // jnp.where(pres, hd, 0)
        den += pr;
      }
      float mean = num / fmaxf(den, 1.0f);
      if (den > 0.0f) { tot += mean; nv++; }
    }
    float res = tot / fmaxf((float)nv, 1.0f);
    out[0] = (nv > 0) ? res : __builtin_inff();
  }
}

// ---------------------------------------------------------------------------
extern "C" void kernel_launch(void* const* d_in, const int* in_sizes, int n_in,
                              void* d_out, int out_size, void* d_ws, size_t ws_size,
                              hipStream_t stream) {
  (void)in_sizes; (void)n_in; (void)out_size; (void)ws_size;
  const float* pred = (const float*)d_in[0];
  const float* targ = (const float*)d_in[1];
  uint8_t* ws = (uint8_t*)d_ws;
  uint8_t* pl = ws;                                   // [NB][NPIX] u8
  uint8_t* tl = ws + (size_t)NB * NPIX;               // [NB][NPIX] u8
  float* hd   = (float*)(ws + (size_t)2 * NB * NPIX); // 52 floats
  float* pres = hd + 64;                              // 52 floats

  labels_kernel<<<dim3((NB * NPIX) / 256), dim3(256), 0, stream>>>(pred, targ, pl, tl);
  hd_pair_kernel<<<dim3(NCLS, NB), dim3(256), sizeof(Smem), stream>>>(pl, tl, hd, pres);
  finalize_kernel<<<dim3(1), dim3(32), 0, stream>>>(hd, pres, (float*)d_out);
}
